// NERRelationModel_34136400069114
// MI455X (gfx1250) — matrix operand order repaired and checked by reading, not compiled, not run on hardware
//
#include <hip/hip_runtime.h>

typedef __attribute__((ext_vector_type(16))) __bf16 v16bf;
typedef __attribute__((ext_vector_type(8)))  __bf16 v8bf;
typedef __attribute__((ext_vector_type(8)))  float  v8f;
typedef unsigned int u32x4 __attribute__((ext_vector_type(4)));
typedef int          i32x4 __attribute__((ext_vector_type(4)));
typedef int          i32x8 __attribute__((ext_vector_type(8)));

#if __has_builtin(__builtin_amdgcn_tensor_load_to_lds)
#define HAVE_TDM 1
#else
#define HAVE_TDM 0
#endif

// ---------------------------------------------------------------- helpers
static __device__ __forceinline__ __bf16 f2bf(float f) {
    return (__bf16)f;                 // native RNE fptrunc -> hw bf16 cvt
}
static __device__ __forceinline__ float warp_sum(float v) {
    #pragma unroll
    for (int m = 16; m > 0; m >>= 1) v += __shfl_xor(v, m, 32);
    return v;
}
static __device__ __forceinline__ float softplusf(float x) {
    return fmaxf(x, 0.f) + log1pf(expf(-fabsf(x)));
}

#if HAVE_TDM
// 2D TDM load: global (tile1 rows x tile0 f32 elements, row stride stride0 elements)
// -> LDS at byte offset lds_off, packed row-major tile1 x tile0.
// Descriptor packing per CDNA5 ISA section 8.3/8.4 (D# group 0 + group 1).
static __device__ __forceinline__ void tdm_load_2d(unsigned lds_off, const void* gptr,
                                                   unsigned dim0, unsigned dim1,
                                                   unsigned tile0, unsigned tile1,
                                                   unsigned stride0) {
    unsigned long long ga = (unsigned long long)(uintptr_t)gptr;
    u32x4 g0;
    g0[0] = 1u;                                            // count=1, user descriptor
    g0[1] = lds_off;                                       // lds_addr (bytes)
    g0[2] = (unsigned)(ga & 0xFFFFFFFFu);                  // global_addr[31:0]
    g0[3] = (unsigned)((ga >> 32) & 0x01FFFFFFu) | (2u << 30); // addr[56:32] | type=2
    i32x8 g1;
    g1[0] = (int)(2u << 16);                               // data_size = 4 bytes
    g1[1] = (int)((dim0 & 0xFFFFu) << 16);                 // tensor_dim0[15:0]
    g1[2] = (int)(((dim0 >> 16) & 0xFFFFu) | ((dim1 & 0xFFFFu) << 16));
    g1[3] = (int)(((dim1 >> 16) & 0xFFFFu) | ((tile0 & 0xFFFFu) << 16)); // tile_dim0
    g1[4] = (int)(tile1 & 0xFFFFu);                        // tile_dim1 (tile_dim2=0)
    g1[5] = (int)stride0;                                  // tensor_dim0_stride[31:0]
    g1[6] = 0;                                             // stride0 hi | stride1 lo
    g1[7] = 0;
    i32x4 z4 = {0, 0, 0, 0};
#if defined(__clang_major__) && __clang_major__ >= 23
    i32x8 z8 = {0, 0, 0, 0, 0, 0, 0, 0};
    __builtin_amdgcn_tensor_load_to_lds(g0, g1, z4, z4, z8, 0);
#else
    __builtin_amdgcn_tensor_load_to_lds(g0, g1, z4, z4, 0);
#endif
}
#endif

// ---------------------------------------------------------------- tiled WMMA GEMM
// C[M,N] = act( A[M,K] @ B[K,N] + bias[N] ); f32 in/out, bf16 WMMA compute.
// Block = 256 threads (8 waves), 64x64 macro-tile. A tile arrives via TDM
// (tensor_load_to_lds) as f32; B tile manually transposed to bf16 in LDS.
// Requires: M%64==0, N%64==0, K%32==0. Dynamic LDS: 8192 (A f32) + 4096 (B bf16).
__global__ void wmma_gemm_tiled(const float* __restrict__ A,
                                const float* __restrict__ B,
                                const float* __restrict__ bias,
                                float* __restrict__ C,
                                int M, int N, int K, int act) {
    extern __shared__ char smem[];
    float*  AbufF = (float*)smem;             // 64 rows x 32 K, f32
    __bf16* Bt    = (__bf16*)(smem + 8192);   // 64 cols x 32 K, bf16 (transposed)

    int nBlk = N >> 6;
    int mB = blockIdx.x / nBlk;
    int nB = blockIdx.x % nBlk;
    int t    = threadIdx.x;
    int wid  = t >> 5;
    int lane = t & 31;
    int half = lane >> 4;
    int l16  = lane & 15;
    int mSub = wid >> 1;
    int nSub = wid & 1;

    int bk = t >> 3, bseg = t & 7;            // B fill: 32 k-rows x 8 col-chunks
#if !HAVE_TDM
    int arow = t >> 2, aq = t & 3;            // A fill fallback
#endif
#if HAVE_TDM
    unsigned ldsA = (unsigned)(unsigned long long)(uintptr_t)(void*)AbufF;
#endif

    v8f acc0 = {}, acc1 = {};

    for (int k0 = 0; k0 < K; k0 += 32) {
        // ---- A tile 64x32 f32 via Tensor Data Mover (one issue per block) ----
#if HAVE_TDM
        if (wid == 0)
            tdm_load_2d(ldsA, A + (long)(mB * 64) * K + k0,
                        /*dim0*/32u, /*dim1*/64u, /*tile0*/32u, /*tile1*/64u,
                        (unsigned)K);
#else
        {
            const float* ap = A + (long)(mB * 64 + arow) * K + k0 + aq * 8;
            *(float4*)&AbufF[arow * 32 + aq * 8]     = *(const float4*)ap;
            *(float4*)&AbufF[arow * 32 + aq * 8 + 4] = *(const float4*)(ap + 4);
        }
#endif
        // ---- B tile transposed to bf16 (overlaps the TDM DMA) ----
        {
            const float* bp = B + (long)(k0 + bk) * N + nB * 64 + bseg * 8;
            float4 g0 = *(const float4*)bp;
            float4 g1 = *(const float4*)(bp + 4);
            float vals[8] = { g0.x, g0.y, g0.z, g0.w, g1.x, g1.y, g1.z, g1.w };
            #pragma unroll
            for (int u = 0; u < 8; ++u)
                Bt[(bseg * 8 + u) * 32 + bk] = f2bf(vals[u]);
        }
        if (k0 + 32 < K)                      // global_prefetch_b8 next B tile
            __builtin_prefetch(B + (long)(k0 + 32 + bk) * N + nB * 64 + bseg * 8, 0, 0);
#if HAVE_TDM
        if (wid == 0) __builtin_amdgcn_s_wait_tensorcnt(0);
#endif
        __syncthreads();

        // ---- A fragment (f32 LDS -> bf16): element e -> K = half*16 + e
        int m = mSub * 16 + l16;
        v16bf a;
        #pragma unroll
        for (int e = 0; e < 8; ++e) {
            a[e]     = f2bf(AbufF[m * 32 + half * 16 + e]);
            a[e + 8] = f2bf(AbufF[m * 32 + half * 16 + 8 + e]);
        }
        // ---- B fragments: lane col c, elements K = half*16 + e (contiguous in Bt)
        int c0 = nSub * 32 + l16;
        v16bf b0 = *(const v16bf*)&Bt[c0 * 32 + half * 16];
        v16bf b1 = *(const v16bf*)&Bt[(c0 + 16) * 32 + half * 16];

        acc0 = __builtin_amdgcn_wmma_f32_16x16x32_bf16(false, a, false, b0,
                                                       (short)0, acc0, false, false);
        acc1 = __builtin_amdgcn_wmma_f32_16x16x32_bf16(false, a, false, b1,
                                                       (short)0, acc1, false, false);
        __syncthreads();
    }

    int ncol0 = nB * 64 + nSub * 32 + l16;
    float bv0 = bias ? bias[ncol0] : 0.f;
    float bv1 = bias ? bias[ncol0 + 16] : 0.f;
    #pragma unroll
    for (int v = 0; v < 8; ++v) {
        int row = mB * 64 + mSub * 16 + v + half * 8;
        float x0 = acc0[v] + bv0;
        float x1 = acc1[v] + bv1;
        if (act == 1) { x0 = fmaxf(x0, 0.f); x1 = fmaxf(x1, 0.f); }
        C[(long)row * N + ncol0]      = x0;
        C[(long)row * N + ncol0 + 16] = x1;
    }
}

// ---------------------------------------------------------------- simple WMMA GEMM (edge N)
__global__ void wmma_gemm_bf16(const float* __restrict__ A,
                               const float* __restrict__ B,
                               const float* __restrict__ bias,
                               float* __restrict__ C,
                               int M, int N, int K, int act) {
    int wave = blockIdx.x * (blockDim.x >> 5) + (threadIdx.x >> 5);
    int lane = threadIdx.x & 31;
    int nT = (N + 15) >> 4;
    int mT = (M + 15) >> 4;
    if (wave >= mT * nT) return;
    int mTile = wave / nT, nTile = wave % nT;
    int half  = lane >> 4;
    int l16   = lane & 15;
    int mrow  = mTile * 16 + l16;
    int ncol  = nTile * 16 + l16;
    bool colOK = (ncol < N);
    int ncolC = colOK ? ncol : (N - 1);
    v8f acc = {};
    for (int k0 = 0; k0 < K; k0 += 32) {
        v16bf a, b;
        #pragma unroll
        for (int e = 0; e < 16; ++e) {
            int kk = half * 16 + e;
            a[e] = f2bf(A[(long)mrow * K + k0 + kk]);
        }
        #pragma unroll
        for (int e = 0; e < 16; ++e) {
            int kk = half * 16 + e;
            float bv = B[(long)(k0 + kk) * N + ncolC];
            b[e] = f2bf(colOK ? bv : 0.f);
        }
        acc = __builtin_amdgcn_wmma_f32_16x16x32_bf16(
                  false, a, false, b, (short)0, acc, false, false);
    }
    float bv = (bias && colOK) ? bias[ncol] : 0.f;
    #pragma unroll
    for (int v = 0; v < 8; ++v) {
        int row = mTile * 16 + v + half * 8;
        if (row < M && colOK) {
            float val = acc[v] + bv;
            if (act == 1) val = fmaxf(val, 0.f);
            C[(long)row * N + ncol] = val;
        }
    }
}

// ---------------------------------------------------------------- accumulator init
__global__ void init_k(float* acc) {
    if (threadIdx.x < 16) acc[threadIdx.x] = 0.f;
}

// ---------------------------------------------------------------- CRF NLL (per batch)
__global__ void crf_k(const float* __restrict__ em_all, const int* __restrict__ tags_all,
                      const int* __restrict__ mask_all, const float* __restrict__ st,
                      const float* __restrict__ tr, const float* __restrict__ et,
                      float* loss_acc) {
    const int L = 11, S = 512;
    int b = blockIdx.x;
    const float* em  = em_all  + (long)b * S * L;
    const int*  tags = tags_all + b * S;
    const int*  mk   = mask_all + b * S;
    __shared__ float alpha[L], nalpha[L];
    __shared__ float red[128];
    __shared__ int  redi[128];
    int t = threadIdx.x;
    if (t < L) alpha[t] = st[t] + em[t];
    __syncthreads();
    for (int s = 1; s < S; ++s) {
        if (t < L) {
            float mx = -1e30f;
            for (int p = 0; p < L; ++p) mx = fmaxf(mx, alpha[p] + tr[p * L + t]);
            float sm = 0.f;
            for (int p = 0; p < L; ++p) sm += expf(alpha[p] + tr[p * L + t] - mx);
            float nv = mx + logf(sm) + em[s * L + t];
            nalpha[t] = (mk[s] > 0) ? nv : alpha[t];
        }
        __syncthreads();
        if (t < L) alpha[t] = nalpha[t];
        __syncthreads();
    }
    float ns = 0.f; int mc = 0;
    for (int s = t; s < S; s += blockDim.x) {
        mc += (mk[s] != 0);
        if (s >= 1 && mk[s] > 0)
            ns += tr[tags[s - 1] * L + tags[s]] + em[s * L + tags[s]];
    }
    red[t] = ns; redi[t] = mc;
    __syncthreads();
    for (int o = blockDim.x >> 1; o > 0; o >>= 1) {
        if (t < o) { red[t] += red[t + o]; redi[t] += redi[t + o]; }
        __syncthreads();
    }
    if (t == 0) {
        int seqlen = redi[0];
        float num = red[0] + st[tags[0]] + em[tags[0]] + et[tags[seqlen - 1]];
        float mx = -1e30f;
        for (int c = 0; c < L; ++c) mx = fmaxf(mx, alpha[c] + et[c]);
        float sm = 0.f;
        for (int c = 0; c < L; ++c) sm += expf(alpha[c] + et[c] - mx);
        float logZ = mx + logf(sm);
        atomicAdd(loss_acc, (logZ - num) * (1.f / 32.f));
    }
}

// ---------------------------------------------------------------- entity mean pool
__global__ void ent_pool(const float* __restrict__ seq, const int* __restrict__ starts,
                         const int* __restrict__ lens, float* __restrict__ ent) {
    int be = blockIdx.x;
    int b = be >> 5;
    int st = starts[be]; if (st > 511) st = 511;
    int en = st + lens[be]; if (en > 511) en = 511;
    float inv = 1.f / (float)(en - st + 1);
    for (int h = threadIdx.x; h < 768; h += blockDim.x) {
        float s = 0.f;
        for (int p = st; p <= en; ++p) s += seq[((long)b * 512 + p) * 768 + h];
        ent[(long)be * 768 + h] = s * inv;
    }
}

// ---------------------------------------------------------------- GAT per-node scores
__global__ void gat_s(const float* __restrict__ h, const float* __restrict__ asrc,
                      const float* __restrict__ adst, float* __restrict__ ssrc,
                      float* __restrict__ sdst, int rows, int Kh, int Dh) {
    int idx = blockIdx.x * blockDim.x + threadIdx.x;
    if (idx >= rows * Kh) return;
    int row = idx / Kh, k = idx % Kh;
    const float* hp = h + ((long)row * Kh + k) * Dh;
    float s1 = 0.f, s2 = 0.f;
    for (int d = 0; d < Dh; ++d) {
        float v = hp[d];
        s1 += v * asrc[k * Dh + d];
        s2 += v * adst[k * Dh + d];
    }
    ssrc[idx] = s1; sdst[idx] = s2;
}

// ---------------------------------------------------------------- GAT softmax+aggregate
__global__ void gat_out(const float* __restrict__ h, const float* __restrict__ ssrc,
                        const float* __restrict__ sdst, const float* __restrict__ bias,
                        float* __restrict__ out, int Kh, int Dh) {
    const int E = 32;
    int idx = blockIdx.x;
    int k = idx % Kh;
    int bi = idx / Kh;
    int b = bi >> 5;
    __shared__ float sc[E];
    int t = threadIdx.x;
    if (t < E) {
        float v = sdst[bi * Kh + k] + ssrc[(b * E + t) * Kh + k];
        sc[t] = (v > 0.f) ? v : 0.2f * v;
    }
    __syncthreads();
    float mx = -1e30f;
    for (int j = 0; j < E; ++j) mx = fmaxf(mx, sc[j]);
    float wsum = 0.f, acc = 0.f;
    for (int j = 0; j < E; ++j) {
        float w = expf(sc[j] - mx);
        wsum += w;
        acc += w * h[((long)(b * E + j) * Kh + k) * Dh + t];
    }
    out[((long)bi * Kh + k) * Dh + t] = acc / wsum + bias[k * Dh + t];
}

// ---------------------------------------------------------------- layernorm + ELU
__global__ void ln_elu(const float* __restrict__ x, const float* __restrict__ g,
                       const float* __restrict__ be, float* __restrict__ y, int F) {
    int row = blockIdx.x;
    const float* xp = x + (long)row * F;
    __shared__ float rs[256], rq[256];
    float s = 0.f, q = 0.f;
    for (int f = threadIdx.x; f < F; f += blockDim.x) {
        float v = xp[f]; s += v; q += v * v;
    }
    rs[threadIdx.x] = s; rq[threadIdx.x] = q;
    __syncthreads();
    for (int o = blockDim.x >> 1; o > 0; o >>= 1) {
        if (threadIdx.x < o) { rs[threadIdx.x] += rs[threadIdx.x + o];
                               rq[threadIdx.x] += rq[threadIdx.x + o]; }
        __syncthreads();
    }
    float mu  = rs[0] / (float)F;
    float var = rq[0] / (float)F - mu * mu;
    float inv = rsqrtf(var + 1e-5f);
    for (int f = threadIdx.x; f < F; f += blockDim.x) {
        float v = (xp[f] - mu) * inv * g[f] + be[f];
        y[(long)row * F + f] = (v > 0.f) ? v : expm1f(v);
    }
}

// ---------------------------------------------------------------- pack Wr1 -> (256 x 2560)
__global__ void pack_k(const float* __restrict__ Wr1, float* __restrict__ Bp) {
    int idx = blockIdx.x * blockDim.x + threadIdx.x;
    if (idx >= 256 * 2560) return;
    int d = idx / 2560, c = idx % 2560;
    int r = c / 512, part = (c >> 8) & 1, h = c & 255;
    Bp[idx] = Wr1[((long)r * 512 + part * 256 + d) * 256 + h];
}

// ---------------------------------------------------------------- pos_r reduction
__global__ void posr_k(const int* __restrict__ tgt, float* posr) {
    int idx = blockIdx.x * blockDim.x + threadIdx.x;
    if (idx >= 32 * 5 * 32 * 32) return;
    int j = idx & 31, i = (idx >> 5) & 31;
    if (i == j) return;
    int r = (idx >> 10) % 5;
    if (tgt[idx]) atomicAdd(&posr[r], (float)tgt[idx]);
}

// ---------------------------------------------------------------- fused pair: LN+relu+dot+BCE
__global__ void pair_k(const float* __restrict__ hij, const float* __restrict__ br1,
                       const float* __restrict__ g, const float* __restrict__ be,
                       const float* __restrict__ Wr2, const float* __restrict__ br2,
                       const int* __restrict__ tgt, const float* __restrict__ posr,
                       float* __restrict__ rel_out, float* bce_acc) {
    int wave = blockIdx.x * (blockDim.x >> 5) + (threadIdx.x >> 5);
    int lane = threadIdx.x & 31;
    const int nC = 32 * 5 * 32 * 32;
    if (wave >= nC) return;
    int j = wave & 31, i = (wave >> 5) & 31, r = (wave >> 10) % 5, b = wave / 5120;
    const float* hi  = hij + ((long)(b * 32 + i) * 2560 + r * 512);
    const float* hjv = hij + ((long)(b * 32 + j) * 2560 + r * 512 + 256);
    float pre[8]; float s = 0.f, q = 0.f;
    #pragma unroll
    for (int u = 0; u < 8; ++u) {
        int h = lane + 32 * u;
        float v = hi[h] + hjv[h] + br1[r * 256 + h];
        pre[u] = v; s += v; q += v * v;
    }
    s = warp_sum(s); q = warp_sum(q);
    float mu  = s * (1.f / 256.f);
    float var = q * (1.f / 256.f) - mu * mu;
    float inv = rsqrtf(var + 1e-5f);
    float dot = 0.f;
    #pragma unroll
    for (int u = 0; u < 8; ++u) {
        int h = lane + 32 * u;
        float v = (pre[u] - mu) * inv * g[r * 256 + h] + be[r * 256 + h];
        v = fmaxf(v, 0.f);
        dot += v * Wr2[r * 256 + h];
    }
    dot = warp_sum(dot);
    float z = dot + br2[r];
    if (lane == 0) {
        rel_out[wave] = z;
        if (i != j) {
            float tg = (float)tgt[wave];
            float pw = fmaxf(1.f, 31744.f / (posr[r] + 1e-6f)) * 5.f;
            float bce = pw * tg * softplusf(-z) + (1.f - tg) * softplusf(z);
            atomicAdd(&bce_acc[r], bce);
        }
    }
}

// ---------------------------------------------------------------- finalize total loss
__global__ void fin_k(const float* acc, float* out_loss) {
    float rl = 0.f;
    for (int r = 0; r < 5; ++r) rl += acc[5 + r] * (1.f / 31744.f);
    *out_loss = acc[10] + rl;
}

// ================================================================ launcher
extern "C" void kernel_launch(void* const* d_in, const int* in_sizes, int n_in,
                              void* d_out, int out_size, void* d_ws, size_t ws_size,
                              hipStream_t stream) {
    const float* seq  = (const float*)d_in[0];
    const int*  amask = (const int*)d_in[1];
    const int*  tags  = (const int*)d_in[2];
    const int*  est   = (const int*)d_in[3];
    const int*  elen  = (const int*)d_in[4];
    const int*  rtgt  = (const int*)d_in[5];
    const float* W1   = (const float*)d_in[6];
    const float* b1   = (const float*)d_in[7];
    const float* W2   = (const float*)d_in[8];
    const float* b2   = (const float*)d_in[9];
    const float* stt  = (const float*)d_in[10];
    const float* trn  = (const float*)d_in[11];
    const float* ett  = (const float*)d_in[12];
    const float* Wg1  = (const float*)d_in[13];
    const float* as1  = (const float*)d_in[14];
    const float* ad1  = (const float*)d_in[15];
    const float* bg1  = (const float*)d_in[16];
    const float* g1g  = (const float*)d_in[17];
    const float* g1b  = (const float*)d_in[18];
    const float* Wg2  = (const float*)d_in[19];
    const float* as2  = (const float*)d_in[20];
    const float* ad2  = (const float*)d_in[21];
    const float* bg2  = (const float*)d_in[22];
    const float* g2g  = (const float*)d_in[23];
    const float* g2b  = (const float*)d_in[24];
    const float* Wr1  = (const float*)d_in[25];
    const float* br1  = (const float*)d_in[26];
    const float* rlg  = (const float*)d_in[27];
    const float* rlb  = (const float*)d_in[28];
    const float* Wr2  = (const float*)d_in[29];
    const float* br2  = (const float*)d_in[30];

    float* out = (float*)d_out;
    float* ner = out;               // 16384 x 11
    float* rel = out + 180224;      // 163840
    float* tot = out + 344064;

    float* ws  = (float*)d_ws;
    float* hid = ws;                  // 16384*256
    float* ent = hid + 4194304;       // 1024*768
    float* h1  = ent + 786432;        // 1024*512
    float* g1o = h1  + 524288;        // 1024*512
    float* x1  = g1o + 524288;        // 1024*512
    float* h2  = x1  + 524288;        // 1024*256
    float* g2o = h2  + 262144;        // 1024*256
    float* x2  = g2o + 262144;        // 1024*256
    float* ss1 = x2  + 262144;        // 4096
    float* sd1 = ss1 + 4096;          // 4096
    float* ss2 = sd1 + 4096;          // 1024
    float* sd2 = ss2 + 1024;          // 1024
    float* Bp  = sd2 + 1024;          // 256*2560
    float* hij = Bp  + 655360;        // 1024*2560
    float* acc = hij + 2621440;       // [0..4]=pos_r [5..9]=bce [10]=ner_loss

    auto gemmT = [&](const float* A, const float* Bm, const float* bias, float* C,
                     int M, int N, int K, int act) {
        int blocks = (M / 64) * (N / 64);
        wmma_gemm_tiled<<<blocks, 256, 12288, stream>>>(A, Bm, bias, C, M, N, K, act);
    };

    init_k<<<1, 32, 0, stream>>>(acc);

    // NER head
    gemmT(seq, W1, b1, hid, 16384, 256, 768, 1);       // relu(seq@W1+b1)
    {   // hid @ W2 + b2 -> ner_logits (N=11 edge case: simple WMMA kernel)
        int mT = (16384 + 15) / 16, nT = (11 + 15) / 16;
        int blocks = (mT * nT + 7) / 8;
        wmma_gemm_bf16<<<blocks, 256, 0, stream>>>(hid, W2, b2, ner, 16384, 11, 256, 0);
    }
    crf_k<<<32, 128, 0, stream>>>(ner, tags, amask, stt, trn, ett, acc + 10);

    // entity pooling + GAT layer 1 (4 heads x 128)
    ent_pool<<<1024, 256, 0, stream>>>(seq, est, elen, ent);
    gemmT(ent, Wg1, nullptr, h1, 1024, 512, 768, 0);
    gat_s<<<16, 256, 0, stream>>>(h1, as1, ad1, ss1, sd1, 1024, 4, 128);
    gat_out<<<4096, 128, 0, stream>>>(h1, ss1, sd1, bg1, g1o, 4, 128);
    ln_elu<<<1024, 256, 0, stream>>>(g1o, g1g, g1b, x1, 512);

    // GAT layer 2 (1 head x 256)
    gemmT(x1, Wg2, nullptr, h2, 1024, 256, 512, 0);
    gat_s<<<4, 256, 0, stream>>>(h2, as2, ad2, ss2, sd2, 1024, 1, 256);
    gat_out<<<1024, 256, 0, stream>>>(h2, ss2, sd2, bg2, g2o, 1, 256);
    ln_elu<<<1024, 256, 0, stream>>>(g2o, g2g, g2b, x2, 256);

    // relation head: pack Wr1, fused hi/hj GEMM, fused pair kernel
    pack_k<<<2560, 256, 0, stream>>>(Wr1, Bp);
    gemmT(x2, Bp, nullptr, hij, 1024, 2560, 256, 0);
    posr_k<<<640, 256, 0, stream>>>(rtgt, acc);
    pair_k<<<20480, 256, 0, stream>>>(hij, br1, rlg, rlb, Wr2, br2, rtgt, acc, rel, acc + 5);

    fin_k<<<1, 1, 0, stream>>>(acc, tot);
}
